// ConvGraphNet_70428873720539
// MI455X (gfx1250) — matrix operand order repaired and checked by reading, not compiled, not run on hardware
//
#include <hip/hip_runtime.h>

typedef __attribute__((ext_vector_type(16))) __bf16 v16bf;
typedef __attribute__((ext_vector_type(8)))  float  v8f;

struct __align__(16) U4 { unsigned int x, y, z, w; };
union Frag16 { U4 u[2]; v16bf v; };

#define NODES 50000
#define EDGES 400000
#define MPAD  50048   // nodes padded to multiple of 64

__device__ __forceinline__ unsigned short f2bf(float f) {
  unsigned int u = __float_as_uint(f);
  u += 0x7FFFu + ((u >> 16) & 1u);          // round-to-nearest-even
  return (unsigned short)(u >> 16);
}
__device__ __forceinline__ float bf2f(unsigned int h) {
  return __uint_as_float(h << 16);
}

// ---------------- degree / norm ----------------
__global__ void k_deg_init(float* deg) {
  int i = blockIdx.x * blockDim.x + threadIdx.x;
  if (i < MPAD) deg[i] = (i < NODES) ? 1.0f : 0.0f;   // self-loop counts 1
}
__global__ void k_deg_accum(const int* __restrict__ dst, float* deg) {
  int e = blockIdx.x * blockDim.x + threadIdx.x;
  if (e < EDGES) atomicAdd(&deg[dst[e]], 1.0f);
}
__global__ void k_dinv(float* deg) {
  int i = blockIdx.x * blockDim.x + threadIdx.x;
  if (i < MPAD) { float d = deg[i]; deg[i] = d > 0.0f ? rsqrtf(d) : 0.0f; }
}
__global__ void k_norm(const int* __restrict__ src, const int* __restrict__ dst,
                       const float* __restrict__ dinv, float* __restrict__ nrm) {
  int e = blockIdx.x * blockDim.x + threadIdx.x;
  if (e < EDGES) nrm[e] = dinv[src[e]] * dinv[dst[e]];
}

// ---------------- f32 -> bf16 input conversion (padded rows zeroed) ----------------
__global__ void k_x_to_bf(const float* __restrict__ x, unsigned short* __restrict__ h) {
  int idx = blockIdx.x * blockDim.x + threadIdx.x;     // one thread per 4 elems
  const int total = MPAD * (128 / 4);
  if (idx >= total) return;
  int i = idx >> 5, c = idx & 31;
  uint2 o;
  if (i < NODES) {
    const float4 v = *(const float4*)(x + (long)i * 128 + c * 4);
    o.x = (unsigned)f2bf(v.x) | ((unsigned)f2bf(v.y) << 16);
    o.y = (unsigned)f2bf(v.z) | ((unsigned)f2bf(v.w) << 16);
  } else { o.x = 0u; o.y = 0u; }
  *(uint2*)(h + (long)i * 128 + c * 4) = o;
}

// ---------------- pack W[K,N] f32 into WMMA B-fragment layout (bf16) ----------------
// Layout index: ((nt*KT + kt)*32 + lane)*16 + half
//   lane<16 : col = nt*16+lane,    K = kt*32 + half + (half>=8 ? 8 : 0)     (half 0..15)
//   lane>=16: col = nt*16+lane-16, K = kt*32 + 8 + half + (half>=8 ? 8 : 0)
__global__ void k_pack_w(const float* __restrict__ W, unsigned short* __restrict__ Wp,
                         int K, int N) {
  int idx = blockIdx.x * blockDim.x + threadIdx.x;
  if (idx >= K * N) return;
  int h    = idx & 15;
  int lane = (idx >> 4) & 31;
  int rest = idx >> 9;
  int KT   = K >> 5;
  int kt   = rest % KT;
  int nt   = rest / KT;
  int col  = nt * 16 + (lane & 15);
  int krow = kt * 32 + ((lane & 16) ? 8 : 0) + h + ((h & 8) ? 8 : 0);
  Wp[idx] = f2bf(W[(long)krow * N + col]);
}

// ---------------- WMMA GEMM: C[MPAD,N](bf16) = A[MPAD,K](bf16) @ Wpacked ----------------
// One wave -> 64x64 output block = 4x4 tiles of 16x16, K in steps of 32.
__global__ void __launch_bounds__(128)
k_gemm_wmma(const unsigned short* __restrict__ A, const unsigned short* __restrict__ Bp,
            unsigned short* __restrict__ C, int N, int K, int mBlocks, int totalTiles) {
  const int lane = threadIdx.x & 31;
  const int tile = blockIdx.x * 4 + (threadIdx.x >> 5);
  if (tile >= totalTiles) return;               // wave-uniform: EXEC stays all-ones
  const int m0 = (tile % mBlocks) * 64;
  const int n0 = (tile / mBlocks) * 64;
  const int KT = K >> 5;

  v8f acc[4][4];
  #pragma unroll
  for (int mi = 0; mi < 4; ++mi)
    #pragma unroll
    for (int ni = 0; ni < 4; ++ni)
      #pragma unroll
      for (int r = 0; r < 8; ++r) acc[mi][ni][r] = 0.0f;

  const unsigned short* pA[4];
  const unsigned short* pB[4];
  #pragma unroll
  for (int mi = 0; mi < 4; ++mi)   // A layout: row=m0+mi*16+(lane&15), K base b=(lane>=16)?8:0
    pA[mi] = A + (long)(m0 + mi * 16 + (lane & 15)) * K + ((lane >> 4) << 3);
  #pragma unroll
  for (int ni = 0; ni < 4; ++ni)
    pB[ni] = Bp + ((long)((n0 >> 4) + ni) * KT * 32 + lane) * 16;

  for (int kt = 0; kt < KT; ++kt) {
    Frag16 a[4], b[4];
    const int k0 = kt << 5;
    #pragma unroll
    for (int mi = 0; mi < 4; ++mi) {
      const unsigned short* p = pA[mi] + k0;
      a[mi].u[0] = *(const U4*)(p);          // K = b .. b+7
      a[mi].u[1] = *(const U4*)(p + 16);     // K = b+16 .. b+23
      __builtin_prefetch(p + 32, 0, 1);      // global_prefetch_b8 of next K slab
    }
    #pragma unroll
    for (int ni = 0; ni < 4; ++ni) {
      const unsigned short* p = pB[ni] + (long)kt * 512;
      b[ni].u[0] = *(const U4*)(p);
      b[ni].u[1] = *(const U4*)(p + 8);
    }
    #pragma unroll
    for (int mi = 0; mi < 4; ++mi)
      #pragma unroll
      for (int ni = 0; ni < 4; ++ni)
        acc[mi][ni] = __builtin_amdgcn_wmma_f32_16x16x32_bf16(
            false, a[mi].v, false, b[ni].v, (short)0, acc[mi][ni], false, false);
  }

  // C/D layout: lane<16 -> N=lane, M=r ; lane>=16 -> N=lane-16, M=8+r
  const int rowOff = (lane & 16) ? 8 : 0;
  const int col = lane & 15;
  #pragma unroll
  for (int mi = 0; mi < 4; ++mi)
    #pragma unroll
    for (int ni = 0; ni < 4; ++ni)
      #pragma unroll
      for (int r = 0; r < 8; ++r) {
        const int row = m0 + mi * 16 + rowOff + r;
        C[(long)row * N + n0 + ni * 16 + col] = f2bf(acc[mi][ni][r]);
      }
}

// ---------------- agg = self-loop term + bias (pad rows zeroed) ----------------
__global__ void k_init_agg(const unsigned short* __restrict__ t, const float* __restrict__ dinv,
                           const float* __restrict__ bias, float* __restrict__ agg,
                           int N, int cshift /* log2(N/4) */) {
  int idx = blockIdx.x * blockDim.x + threadIdx.x;
  int total = MPAD << cshift;
  if (idx >= total) return;
  int i = idx >> cshift;
  int c = idx & ((1 << cshift) - 1);
  float* out = agg + (long)i * N + c * 4;
  float4 o;
  if (i >= NODES) { o.x = o.y = o.z = o.w = 0.0f; *(float4*)out = o; return; }
  float dv = dinv[i]; float s = dv * dv;
  uint2 v = *(const uint2*)(t + (long)i * N + c * 4);
  int f = c * 4;
  o.x = bf2f(v.x & 0xffffu) * s + bias[f + 0];
  o.y = bf2f(v.x >> 16)     * s + bias[f + 1];
  o.z = bf2f(v.y & 0xffffu) * s + bias[f + 2];
  o.w = bf2f(v.y >> 16)     * s + bias[f + 3];
  *(float4*)out = o;
}

// ---------------- edge scatter: agg[dst] += tmp[src] * norm (8 feats/thread) ----------------
__global__ void k_scatter(const unsigned short* __restrict__ t, const int* __restrict__ src,
                          const int* __restrict__ dst, const float* __restrict__ nrm,
                          float* __restrict__ agg, int N, int cshift /* log2(N/8) */) {
  long tid = (long)blockIdx.x * blockDim.x + threadIdx.x;
  long total = (long)EDGES << cshift;
  if (tid >= total) return;
  int c = (int)(tid & ((1 << cshift) - 1));
  int e = (int)(tid >> cshift);
  int s = src[e], d = dst[e];
  float w = nrm[e];
  const U4 v = *(const U4*)(t + (long)s * N + c * 8);
  float* out = agg + (long)d * N + c * 8;
  atomicAdd(out + 0, bf2f(v.x & 0xffffu) * w);
  atomicAdd(out + 1, bf2f(v.x >> 16)     * w);
  atomicAdd(out + 2, bf2f(v.y & 0xffffu) * w);
  atomicAdd(out + 3, bf2f(v.y >> 16)     * w);
  atomicAdd(out + 4, bf2f(v.z & 0xffffu) * w);
  atomicAdd(out + 5, bf2f(v.z >> 16)     * w);
  atomicAdd(out + 6, bf2f(v.w & 0xffffu) * w);
  atomicAdd(out + 7, bf2f(v.w >> 16)     * w);
}

// ---------------- ReLU + bf16 conversion into next-layer activations ----------------
__global__ void k_relu_bf(const float* __restrict__ agg, unsigned short* __restrict__ h,
                          int N, int cshift /* log2(N/4) */) {
  int idx = blockIdx.x * blockDim.x + threadIdx.x;
  int total = MPAD << cshift;
  if (idx >= total) return;
  int i = idx >> cshift;
  int c = idx & ((1 << cshift) - 1);
  float4 v = *(const float4*)(agg + (long)i * N + c * 4);
  uint2 o;
  o.x = (unsigned)f2bf(fmaxf(v.x, 0.0f)) | ((unsigned)f2bf(fmaxf(v.y, 0.0f)) << 16);
  o.y = (unsigned)f2bf(fmaxf(v.z, 0.0f)) | ((unsigned)f2bf(fmaxf(v.w, 0.0f)) << 16);
  *(uint2*)(h + (long)i * N + c * 4) = o;
}

// ---------------- last layer: 64 -> 1 ----------------
__global__ void k_dot64(const unsigned short* __restrict__ h, const float* __restrict__ W,
                        float* __restrict__ t1) {
  int i = blockIdx.x * blockDim.x + threadIdx.x;
  if (i >= NODES) return;
  const unsigned short* p = h + (long)i * 64;
  float s = 0.0f;
  #pragma unroll
  for (int k = 0; k < 64; ++k) s += bf2f((unsigned int)p[k]) * W[k];
  t1[i] = s;
}
__global__ void k_init1(const float* __restrict__ t1, const float* __restrict__ dinv,
                        const float* __restrict__ b, float* __restrict__ a1) {
  int i = blockIdx.x * blockDim.x + threadIdx.x;
  if (i >= NODES) return;
  float dv = dinv[i];
  a1[i] = t1[i] * dv * dv + b[0];
}
__global__ void k_scatter1(const float* __restrict__ t1, const int* __restrict__ src,
                           const int* __restrict__ dst, const float* __restrict__ nrm,
                           float* __restrict__ a1) {
  int e = blockIdx.x * blockDim.x + threadIdx.x;
  if (e >= EDGES) return;
  atomicAdd(&a1[dst[e]], t1[src[e]] * nrm[e]);
}
__global__ void k_copy(const float* __restrict__ a1, float* __restrict__ out) {
  int i = blockIdx.x * blockDim.x + threadIdx.x;
  if (i < NODES) out[i] = a1[i];
}

// ---------------- orchestration ----------------
extern "C" void kernel_launch(void* const* d_in, const int* in_sizes, int n_in,
                              void* d_out, int out_size, void* d_ws, size_t ws_size,
                              hipStream_t stream) {
  (void)in_sizes; (void)n_in; (void)out_size; (void)ws_size;
  static const int dims[6] = {128, 1024, 512, 256, 64, 1};
  const float* x  = (const float*)d_in[0];
  const int*   ei = (const int*)d_in[1];
  const int* src = ei;
  const int* dst = ei + EDGES;
  const float* W[5]; const float* B[5];
  for (int l = 0; l < 5; ++l) { W[l] = (const float*)d_in[2 + l]; B[l] = (const float*)d_in[7 + l]; }

  char* ws = (char*)d_ws;
  size_t off = 0;
  auto carve = [&](size_t bytes) -> void* {
    off = (off + 255) & ~(size_t)255;
    void* p = ws + off;
    off += bytes;
    return p;
  };
  float* dinv           = (float*)carve((size_t)MPAD * 4);
  float* nrm            = (float*)carve((size_t)EDGES * 4);
  unsigned short* hbf   = (unsigned short*)carve((size_t)MPAD * 1024 * 2);
  unsigned short* tmpbf = (unsigned short*)carve((size_t)MPAD * 1024 * 2);
  float* agg            = (float*)carve((size_t)MPAD * 1024 * 4);
  unsigned short* wp[4];
  for (int l = 0; l < 4; ++l)
    wp[l] = (unsigned short*)carve((size_t)dims[l] * dims[l + 1] * 2);
  float* t1 = (float*)carve((size_t)MPAD * 4);
  float* a1 = (float*)carve((size_t)MPAD * 4);

  const int T = 256;
  k_deg_init <<<(MPAD  + T - 1) / T, T, 0, stream>>>(dinv);
  k_deg_accum<<<(EDGES + T - 1) / T, T, 0, stream>>>(dst, dinv);
  k_dinv     <<<(MPAD  + T - 1) / T, T, 0, stream>>>(dinv);
  k_norm     <<<(EDGES + T - 1) / T, T, 0, stream>>>(src, dst, dinv, nrm);

  for (int l = 0; l < 4; ++l) {
    int n = dims[l] * dims[l + 1];
    k_pack_w<<<(n + T - 1) / T, T, 0, stream>>>(W[l], wp[l], dims[l], dims[l + 1]);
  }
  k_x_to_bf<<<(MPAD * 32 + T - 1) / T, T, 0, stream>>>(x, hbf);

  for (int l = 0; l < 4; ++l) {
    const int K = dims[l], N = dims[l + 1];
    const int mBlocks = MPAD / 64;
    const int totalTiles = mBlocks * (N / 64);
    k_gemm_wmma<<<(totalTiles + 3) / 4, 128, 0, stream>>>(hbf, wp[l], tmpbf, N, K,
                                                          mBlocks, totalTiles);
    const int cs4 = __builtin_ctz((unsigned)(N / 4));
    const int cs8 = __builtin_ctz((unsigned)(N / 8));
    const int tInit = MPAD << cs4;
    k_init_agg<<<(tInit + T - 1) / T, T, 0, stream>>>(tmpbf, dinv, B[l], agg, N, cs4);
    const long tScat = (long)EDGES << cs8;
    k_scatter<<<(unsigned)((tScat + T - 1) / T), T, 0, stream>>>(tmpbf, src, dst, nrm,
                                                                 agg, N, cs8);
    k_relu_bf<<<(tInit + T - 1) / T, T, 0, stream>>>(agg, hbf, N, cs4);
  }

  k_dot64   <<<(NODES + T - 1) / T, T, 0, stream>>>(hbf, W[4], t1);
  k_init1   <<<(NODES + T - 1) / T, T, 0, stream>>>(t1, dinv, B[4], a1);
  k_scatter1<<<(EDGES + T - 1) / T, T, 0, stream>>>(t1, src, dst, nrm, a1);
  k_copy    <<<(NODES + T - 1) / T, T, 0, stream>>>(a1, (float*)d_out);
}
